// MicroGPT_69209103007941
// MI455X (gfx1250) — compile-verified, matrix operand order checked by reading, not run on hardware
//
#include <hip/hip_runtime.h>
#include <math.h>

// ---------------------------------------------------------------------------
// MicroGPT forward for MI455X (gfx1250, wave32, WMMA).
// GEMMs: v_wmma_f32_16x16x32_bf16 + double-buffered async global->LDS staging
// (global_load_async_to_lds_b128 / s_wait_asynccnt).
// ---------------------------------------------------------------------------

typedef __bf16 bf16;
typedef __attribute__((ext_vector_type(16))) __bf16 bf16x16;
typedef __attribute__((ext_vector_type(8)))  float  f32x8;

#define SEQ    2048
#define DIM    1024
#define FF     4096
#define VOC    32000
#define LAYERS 12
#define NBLK   16            // SEQ / 128
#define EPSF   1.1920929e-07f

// ---------------------------------------------------------------------------
// CDNA5 async copy: 16B global -> LDS, tracked by ASYNCcnt (CDNA5 ISA)
// LDS operand = wave-relative LDS byte address = low 32 bits of generic ptr.
// ---------------------------------------------------------------------------
__device__ __forceinline__ void async_copy16(const void* gsrc, void* lds_dst) {
    uint32_t loff = (uint32_t)(uintptr_t)lds_dst;
    uint64_t gaddr = (uint64_t)(uintptr_t)gsrc;
    asm volatile("global_load_async_to_lds_b128 %0, %1, off"
                 :: "v"(loff), "v"(gaddr) : "memory");
}
__device__ __forceinline__ void wait_async0() {
    asm volatile("s_wait_asynccnt 0x0" ::: "memory");
}

// ---------------------------------------------------------------------------
// fp32 -> bf16 conversion (grid-stride)
// ---------------------------------------------------------------------------
__global__ void k_cvt_bf16(const float* __restrict__ src, bf16* __restrict__ dst, size_t n) {
    size_t i = (size_t)blockIdx.x * blockDim.x + threadIdx.x;
    size_t stride = (size_t)gridDim.x * blockDim.x;
    for (; i < n; i += stride) dst[i] = (bf16)src[i];
}

// ---------------------------------------------------------------------------
// doc boundaries: docs = cumsum(inputs == 2); per-128-block lo/hi
// ---------------------------------------------------------------------------
__global__ void k_docs(const int* __restrict__ inputs, int* __restrict__ docs,
                       int* __restrict__ dl, int* __restrict__ dh) {
    int c = 0;
    for (int i = 0; i < SEQ; ++i) { c += (inputs[i] == 2) ? 1 : 0; docs[i] = c; }
    for (int b = 0; b < NBLK; ++b) { dl[b] = docs[b * 128]; dh[b] = docs[b * 128 + 127]; }
}

// ---------------------------------------------------------------------------
// x = x0 = rms_norm(embed[inputs])   (one block per token row, 256 thr)
// ---------------------------------------------------------------------------
__global__ __launch_bounds__(256)
void k_embed(const float* __restrict__ embed, const int* __restrict__ inputs,
             float* __restrict__ x, float* __restrict__ x0) {
    __shared__ float red[256];
    const int t = blockIdx.x;
    const float* e = embed + (size_t)inputs[t] * DIM;
    float v[4]; float ss = 0.f;
#pragma unroll
    for (int j = 0; j < 4; ++j) { v[j] = e[threadIdx.x + j * 256]; ss += v[j] * v[j]; }
    red[threadIdx.x] = ss; __syncthreads();
    for (int s = 128; s > 0; s >>= 1) {
        if (threadIdx.x < s) red[threadIdx.x] += red[threadIdx.x + s];
        __syncthreads();
    }
    const float r = rsqrtf(red[0] / (float)DIM + EPSF);
    const size_t base = (size_t)t * DIM;
#pragma unroll
    for (int j = 0; j < 4; ++j) {
        float o = v[j] * r;
        x[base + threadIdx.x + j * 256]  = o;
        x0[base + threadIdx.x + j * 256] = o;
    }
}

// ---------------------------------------------------------------------------
// x = blk0*(x [+ sw*skip]) + blk1*x0 ; xn = bf16(rms_norm(x))
// ---------------------------------------------------------------------------
__global__ __launch_bounds__(256)
void k_layer_pre(float* __restrict__ x, const float* __restrict__ x0,
                 const float* __restrict__ skip, const float* __restrict__ skip_w, int swi,
                 const float* __restrict__ blk_lam, int layer, bf16* __restrict__ xn) {
    __shared__ float red[256];
    const int t = blockIdx.x;
    const float b0 = blk_lam[2 * layer], b1 = blk_lam[2 * layer + 1];
    const float sw = skip ? skip_w[swi] : 0.f;
    const size_t base = (size_t)t * DIM;
    float v[4]; float ss = 0.f;
#pragma unroll
    for (int j = 0; j < 4; ++j) {
        int d = threadIdx.x + j * 256;
        float xv = x[base + d];
        if (skip) xv += sw * skip[base + d];
        xv = b0 * xv + b1 * x0[base + d];
        v[j] = xv; ss += xv * xv;
    }
    red[threadIdx.x] = ss; __syncthreads();
    for (int s = 128; s > 0; s >>= 1) {
        if (threadIdx.x < s) red[threadIdx.x] += red[threadIdx.x + s];
        __syncthreads();
    }
    const float r = rsqrtf(red[0] / (float)DIM + EPSF);
#pragma unroll
    for (int j = 0; j < 4; ++j) {
        int d = threadIdx.x + j * 256;
        x[base + d]  = v[j];
        xn[base + d] = (bf16)(v[j] * r);
    }
}

// ---------------------------------------------------------------------------
// dst = bf16(rms_norm(src))  (row kernel)
// ---------------------------------------------------------------------------
__global__ __launch_bounds__(256)
void k_rms_bf(const float* __restrict__ src, bf16* __restrict__ dst) {
    __shared__ float red[256];
    const int t = blockIdx.x;
    const size_t base = (size_t)t * DIM;
    float v[4]; float ss = 0.f;
#pragma unroll
    for (int j = 0; j < 4; ++j) { v[j] = src[base + threadIdx.x + j * 256]; ss += v[j] * v[j]; }
    red[threadIdx.x] = ss; __syncthreads();
    for (int s = 128; s > 0; s >>= 1) {
        if (threadIdx.x < s) red[threadIdx.x] += red[threadIdx.x + s];
        __syncthreads();
    }
    const float r = rsqrtf(red[0] / (float)DIM + EPSF);
#pragma unroll
    for (int j = 0; j < 4; ++j) dst[base + threadIdx.x + j * 256] = (bf16)(v[j] * r);
}

// ---------------------------------------------------------------------------
// dst = bf16( rotary( rms_norm(src_row) ) )   (q / k path)
// ---------------------------------------------------------------------------
__global__ __launch_bounds__(256)
void k_rope(const float* __restrict__ src, bf16* __restrict__ dst) {
    __shared__ float sv[DIM];
    __shared__ float red[256];
    const int t = blockIdx.x;
    const size_t base = (size_t)t * DIM;
    float ss = 0.f;
#pragma unroll
    for (int j = 0; j < 4; ++j) {
        int d = threadIdx.x + j * 256;
        float v = src[base + d];
        sv[d] = v; ss += v * v;
    }
    red[threadIdx.x] = ss; __syncthreads();
    for (int s = 128; s > 0; s >>= 1) {
        if (threadIdx.x < s) red[threadIdx.x] += red[threadIdx.x + s];
        __syncthreads();
    }
    const float r = rsqrtf(red[0] / (float)DIM + EPSF);
    const float log1e4_over = 9.210340372f / 512.0f;   // ln(10000)/512
#pragma unroll
    for (int j = 0; j < 4; ++j) {
        int d = threadIdx.x + j * 256;
        int jj = d & 511;
        float inv = expf(-(float)jj * log1e4_over);
        float ang = (float)t * inv;
        float c = cosf(ang), s = sinf(ang);
        float x1 = sv[jj] * r, x2 = sv[jj + 512] * r;
        float o = (d < 512) ? (x1 * c + x2 * s) : (-x1 * s + x2 * c);
        dst[base + d] = (bf16)o;
    }
}

// ---------------------------------------------------------------------------
// v_bf = bf16( al0 * v + al1 * ve_sel[inputs[t]] )
// ---------------------------------------------------------------------------
__global__ void k_vcomb(const float* __restrict__ vsrc, const float* __restrict__ ve_sel,
                        const int* __restrict__ inputs, const float* __restrict__ attn_lam,
                        int layer, bf16* __restrict__ vdst) {
    size_t i = (size_t)blockIdx.x * blockDim.x + threadIdx.x;
    if (i >= (size_t)SEQ * DIM) return;
    const float a0 = attn_lam[2 * layer], a1 = attn_lam[2 * layer + 1];
    int t = (int)(i / DIM), d = (int)(i % DIM);
    float vv = a0 * vsrc[i] + a1 * ve_sel[(size_t)inputs[t] * DIM + d];
    vdst[i] = (bf16)vv;
}

// ---------------------------------------------------------------------------
// masked softmax row kernel: p_bf[t,:] = softmax(scale*scores masked)
// ---------------------------------------------------------------------------
__global__ __launch_bounds__(256)
void k_softmax(const float* __restrict__ scores, const int* __restrict__ docs,
               const int* __restrict__ dl, const int* __restrict__ dh,
               const int* __restrict__ swnb_p, bf16* __restrict__ p) {
    __shared__ float prow[SEQ];
    __shared__ float red[256];
    const int t = blockIdx.x;
    const int swnb = swnb_p[0];
    const int bq = t >> 7;
    const int dt = docs[t];
    const float scale = 0.03125f;           // 1/sqrt(1024)
    const size_t base = (size_t)t * SEQ;
    float sv[8];
    float mx = -3.0e38f;
#pragma unroll
    for (int j = 0; j < 8; ++j) {
        int c = threadIdx.x + j * 256;
        int bk = c >> 7;
        bool blk = (bq >= bk) && (bq - bk < swnb) && (dl[bq] <= dh[bk]) && (dl[bk] <= dh[bq]);
        bool ok = blk && (t >= c) && (dt == docs[c]);
        float s = ok ? scores[base + c] * scale : -INFINITY;
        sv[j] = s; mx = fmaxf(mx, s);
    }
    red[threadIdx.x] = mx; __syncthreads();
    for (int s = 128; s > 0; s >>= 1) {
        if (threadIdx.x < s) red[threadIdx.x] = fmaxf(red[threadIdx.x], red[threadIdx.x + s]);
        __syncthreads();
    }
    mx = red[0]; __syncthreads();
    float sum = 0.f;
#pragma unroll
    for (int j = 0; j < 8; ++j) {
        int c = threadIdx.x + j * 256;
        float e = expf(sv[j] - mx);
        prow[c] = e; sum += e;
    }
    red[threadIdx.x] = sum; __syncthreads();
    for (int s = 128; s > 0; s >>= 1) {
        if (threadIdx.x < s) red[threadIdx.x] += red[threadIdx.x + s];
        __syncthreads();
    }
    const float inv = 1.f / red[0];
#pragma unroll
    for (int j = 0; j < 8; ++j) {
        int c = threadIdx.x + j * 256;
        p[base + c] = (bf16)(prow[c] * inv);
    }
}

// ---------------------------------------------------------------------------
// elementwise helpers
// ---------------------------------------------------------------------------
__global__ void k_add(float* __restrict__ x, const float* __restrict__ y, size_t n) {
    size_t i = (size_t)blockIdx.x * blockDim.x + threadIdx.x;
    size_t stride = (size_t)gridDim.x * blockDim.x;
    for (; i < n; i += stride) x[i] += y[i];
}
__global__ void k_copy(float* __restrict__ dst, const float* __restrict__ src, size_t n) {
    size_t i = (size_t)blockIdx.x * blockDim.x + threadIdx.x;
    size_t stride = (size_t)gridDim.x * blockDim.x;
    for (; i < n; i += stride) dst[i] = src[i];
}
__global__ void k_relusq(const float* __restrict__ src, bf16* __restrict__ dst, size_t n) {
    size_t i = (size_t)blockIdx.x * blockDim.x + threadIdx.x;
    size_t stride = (size_t)gridDim.x * blockDim.x;
    for (; i < n; i += stride) { float r = fmaxf(src[i], 0.f); dst[i] = (bf16)(r * r); }
}
__global__ void k_cap(float* __restrict__ l, size_t n) {
    size_t i = (size_t)blockIdx.x * blockDim.x + threadIdx.x;
    size_t stride = (size_t)gridDim.x * blockDim.x;
    for (; i < n; i += stride) l[i] = 30.f * tanhf(l[i] * (1.f / 30.f));
}

// ---------------------------------------------------------------------------
// per-row cross-entropy on capped logits
// ---------------------------------------------------------------------------
__global__ __launch_bounds__(256)
void k_row_loss(const float* __restrict__ logits, const int* __restrict__ targets,
                float* __restrict__ rowloss) {
    __shared__ float red[256];
    const int t = blockIdx.x;
    const float* row = logits + (size_t)t * VOC;
    float mx = -3.0e38f;
    for (int j = 0; j < 125; ++j) mx = fmaxf(mx, row[threadIdx.x + j * 256]);
    red[threadIdx.x] = mx; __syncthreads();
    for (int s = 128; s > 0; s >>= 1) {
        if (threadIdx.x < s) red[threadIdx.x] = fmaxf(red[threadIdx.x], red[threadIdx.x + s]);
        __syncthreads();
    }
    mx = red[0]; __syncthreads();
    float sum = 0.f;
    for (int j = 0; j < 125; ++j) sum += expf(row[threadIdx.x + j * 256] - mx);
    red[threadIdx.x] = sum; __syncthreads();
    for (int s = 128; s > 0; s >>= 1) {
        if (threadIdx.x < s) red[threadIdx.x] += red[threadIdx.x + s];
        __syncthreads();
    }
    if (threadIdx.x == 0)
        rowloss[t] = -(row[targets[t]] - mx - logf(red[0]));
}

__global__ __launch_bounds__(256)
void k_loss(const float* __restrict__ rowloss, float* __restrict__ out) {
    __shared__ float red[256];
    float s = 0.f;
    for (int j = 0; j < 8; ++j) s += rowloss[threadIdx.x + j * 256];
    red[threadIdx.x] = s; __syncthreads();
    for (int st = 128; st > 0; st >>= 1) {
        if (threadIdx.x < st) red[threadIdx.x] += red[threadIdx.x + st];
        __syncthreads();
    }
    if (threadIdx.x == 0) out[0] = red[0] / (float)SEQ;
}

// ---------------------------------------------------------------------------
// WMMA bf16 GEMM:  C[M,N] = alpha * A[M,K] * op(B)
//   TRANSB=1 : B is [N,K] row-major (x @ W.T pattern, and q @ k.T)
//   TRANSB=0 : B is [K,N] row-major (p @ v pattern)
// 128x128 block tile, BK=32, 8 waves (wave32), wave tile 32x64 = 8 WMMA accs.
// Double-buffered LDS, staged with global_load_async_to_lds_b128 (ASYNCcnt),
// so tile k+1 streams into LDS while the matrix pipe chews tile k.
// ---------------------------------------------------------------------------
__device__ __forceinline__ bf16x16 load_frag(const bf16* base, int ld, int lane) {
    const int r  = lane & 15;
    const int hi = (lane >> 4) & 1;
    const bf16* row = base + r * ld;
    bf16x16 f;
#pragma unroll
    for (int v = 0; v < 8; ++v) {
        int k = ((v & 4) << 2) + ((v & 3) << 1) + (hi << 3);
        f[2 * v]     = row[k];
        f[2 * v + 1] = row[k + 1];
    }
    return f;
}

template <int TRANSB>
__global__ __launch_bounds__(256)
void k_gemm(const bf16* __restrict__ A, const bf16* __restrict__ B,
            float* __restrict__ C, int M, int N, int K, float alpha) {
    constexpr int BM = 128, BN = 128, BK = 32;
    constexpr int LDP = 40;                 // 40 bf16 = 80 B rows (16B aligned)
    __shared__ bf16 As[2][BM][LDP];
    __shared__ bf16 Bs[2][BN][LDP];

    const int tid  = threadIdx.x;
    const int lane = tid & 31;
    const int wave = tid >> 5;
    const int wrow = wave & 3;              // 4 wave-rows of 32
    const int wcol = wave >> 2;             // 2 wave-cols of 64
    const int row0 = blockIdx.y * BM;
    const int col0 = blockIdx.x * BN;

    // stage one 128x32 A tile + B tile at K-offset k0 into buffer `buf`
    auto stage = [&](int buf, int k0) {
        // A tile: 512 chunks of 8 bf16, 2 per thread, async DMA into LDS
#pragma unroll
        for (int i = 0; i < 2; ++i) {
            int cid = tid + i * 256;
            int r   = cid >> 2;             // 0..127
            int kc  = (cid & 3) << 3;       // 0,8,16,24
            async_copy16(A + (size_t)(row0 + r) * K + k0 + kc, &As[buf][r][kc]);
        }
        if constexpr (TRANSB) {
#pragma unroll
            for (int i = 0; i < 2; ++i) {
                int cid = tid + i * 256;
                int n   = cid >> 2;
                int kc  = (cid & 3) << 3;
                async_copy16(B + (size_t)(col0 + n) * K + k0 + kc, &Bs[buf][n][kc]);
            }
        } else {
            // transpose path: batch both global loads, then scatter to LDS
            int kk0 = tid >> 4,          nc0 = (tid & 15) << 3;
            int kk1 = (tid + 256) >> 4,  nc1 = ((tid + 256) & 15) << 3;
            union { uint4 u; bf16 h[8]; } t0, t1;
            t0.u = *(const uint4*)(B + (size_t)(k0 + kk0) * N + col0 + nc0);
            t1.u = *(const uint4*)(B + (size_t)(k0 + kk1) * N + col0 + nc1);
#pragma unroll
            for (int j = 0; j < 8; ++j) Bs[buf][nc0 + j][kk0] = t0.h[j];
#pragma unroll
            for (int j = 0; j < 8; ++j) Bs[buf][nc1 + j][kk1] = t1.h[j];
        }
    };

    f32x8 acc[2][4];
#pragma unroll
    for (int mi = 0; mi < 2; ++mi)
#pragma unroll
        for (int ni = 0; ni < 4; ++ni)
#pragma unroll
            for (int v = 0; v < 8; ++v) acc[mi][ni][v] = 0.f;

    stage(0, 0);
    wait_async0();
    __syncthreads();

    int cur = 0;
    for (int k0 = 0; k0 < K; k0 += BK) {
        const int nxt = cur ^ 1;
        if (k0 + BK < K) stage(nxt, k0 + BK);   // overlap copy with math

        bf16x16 af[2], bfm[4];
#pragma unroll
        for (int mi = 0; mi < 2; ++mi)
            af[mi] = load_frag(&As[cur][wrow * 32 + mi * 16][0], LDP, lane);
#pragma unroll
        for (int ni = 0; ni < 4; ++ni)
            bfm[ni] = load_frag(&Bs[cur][wcol * 64 + ni * 16][0], LDP, lane);

#pragma unroll
        for (int mi = 0; mi < 2; ++mi)
#pragma unroll
            for (int ni = 0; ni < 4; ++ni)
                acc[mi][ni] = __builtin_amdgcn_wmma_f32_16x16x32_bf16(
                    false, af[mi], false, bfm[ni], (short)0, acc[mi][ni], false, false);

        wait_async0();       // my async writes to buf `nxt` have landed
        __syncthreads();     // everyone done reading `cur` / writing `nxt`
        cur = nxt;
    }

    // ---- store C: lanes 0-15 -> M=v, lanes 16-31 -> M=v+8 ; N = lane&15 ----
#pragma unroll
    for (int mi = 0; mi < 2; ++mi) {
#pragma unroll
        for (int ni = 0; ni < 4; ++ni) {
            int rb = row0 + wrow * 32 + mi * 16 + ((lane >> 4) << 3);
            int c  = col0 + wcol * 64 + ni * 16 + (lane & 15);
#pragma unroll
            for (int v = 0; v < 8; ++v)
                C[(size_t)(rb + v) * N + c] = alpha * acc[mi][ni][v];
        }
    }
}

// ---------------------------------------------------------------------------
// host driver
// ---------------------------------------------------------------------------
extern "C" void kernel_launch(void* const* d_in, const int* in_sizes, int n_in,
                              void* d_out, int out_size, void* d_ws, size_t ws_size,
                              hipStream_t stream) {
    (void)in_sizes; (void)n_in; (void)out_size; (void)ws_size;

    const int*   inputs   = (const int*)d_in[0];
    const int*   targets  = (const int*)d_in[1];
    const int*   swnb     = (const int*)d_in[2];
    const float* embed    = (const float*)d_in[3];
    const float* ve       = (const float*)d_in[4];
    const float* Wq       = (const float*)d_in[5];
    const float* Wk       = (const float*)d_in[6];
    const float* Wv       = (const float*)d_in[7];
    const float* Wo       = (const float*)d_in[8];
    const float* attn_lam = (const float*)d_in[9];
    const float* Wfc      = (const float*)d_in[10];
    const float* Wpr      = (const float*)d_in[11];
    const float* blk_lam  = (const float*)d_in[12];
    const float* skip_w   = (const float*)d_in[13];
    const float* lm_head  = (const float*)d_in[14];

    static const int VE_IDX[LAYERS] = {0, 1, 2, 3, 4, 5, 5, 4, 3, 2, 1, 0};

    const size_t TD  = (size_t)SEQ * DIM;
    const size_t TH  = (size_t)SEQ * FF;
    const size_t T2  = (size_t)SEQ * SEQ;
    const size_t WDD = (size_t)LAYERS * DIM * DIM;
    const size_t WFD = (size_t)LAYERS * FF * DIM;
    const size_t LMN = (size_t)VOC * DIM;

    char* p = (char*)d_ws;
    auto alloc = [&](size_t bytes) -> char* {
        char* r = p; p += (bytes + 255) & ~(size_t)255; return r;
    };

    bf16* wq_bf  = (bf16*)alloc(2 * WDD);
    bf16* wk_bf  = (bf16*)alloc(2 * WDD);
    bf16* wv_bf  = (bf16*)alloc(2 * WDD);
    bf16* wo_bf  = (bf16*)alloc(2 * WDD);
    bf16* wfc_bf = (bf16*)alloc(2 * WFD);
    bf16* wpr_bf = (bf16*)alloc(2 * WFD);
    bf16* lm_bf  = (bf16*)alloc(2 * LMN);
    float* x     = (float*)alloc(4 * TD);
    float* x0    = (float*)alloc(4 * TD);
    bf16*  xn_bf = (bf16*)alloc(2 * TD);
    float* qf    = (float*)alloc(4 * TD);
    float* kf    = (float*)alloc(4 * TD);
    float* vf    = (float*)alloc(4 * TD);
    bf16*  qb    = (bf16*)alloc(2 * TD);
    bf16*  kb    = (bf16*)alloc(2 * TD);
    bf16*  vb    = (bf16*)alloc(2 * TD);
    float* sc    = (float*)alloc(4 * T2);
    bf16*  pb    = (bf16*)alloc(2 * T2);
    float* yf    = (float*)alloc(4 * TD);
    bf16*  yb    = (bf16*)alloc(2 * TD);
    float* of    = (float*)alloc(4 * TD);
    float* hf    = (float*)alloc(4 * TH);
    bf16*  hb    = (bf16*)alloc(2 * TH);
    float* mf    = (float*)alloc(4 * TD);
    float* skips = (float*)alloc(4 * 6 * TD);
    int*   docs  = (int*)alloc(4 * SEQ);
    int*   dlp   = (int*)alloc(4 * NBLK);
    int*   dhp   = (int*)alloc(4 * NBLK);
    float* rowls = (float*)alloc(4 * SEQ);

    auto cvt = [&](const float* s, bf16* d, size_t n) {
        int blocks = (int)((n + 256 * 16 - 1) / (256 * 16));
        if (blocks < 1) blocks = 1;
        k_cvt_bf16<<<blocks, 256, 0, stream>>>(s, d, n);
    };
    auto gemm = [&](const bf16* A, const bf16* B, float* Cc,
                    int M, int N, int K, bool transB) {
        dim3 g(N / 128, M / 128), b(256);
        if (transB) k_gemm<1><<<g, b, 0, stream>>>(A, B, Cc, M, N, K, 1.0f);
        else        k_gemm<0><<<g, b, 0, stream>>>(A, B, Cc, M, N, K, 1.0f);
    };

    // weights -> bf16 (every call; deterministic)
    cvt(Wq, wq_bf, WDD);  cvt(Wk, wk_bf, WDD);
    cvt(Wv, wv_bf, WDD);  cvt(Wo, wo_bf, WDD);
    cvt(Wfc, wfc_bf, WFD); cvt(Wpr, wpr_bf, WFD);
    cvt(lm_head, lm_bf, LMN);

    k_docs<<<1, 1, 0, stream>>>(inputs, docs, dlp, dhp);
    k_embed<<<SEQ, 256, 0, stream>>>(embed, inputs, x, x0);

    const int EW = 8192;   // elementwise grid for TD-sized arrays
    for (int l = 0; l < LAYERS; ++l) {
        const float* skip = (l >= 6) ? (skips + (size_t)(11 - l) * TD) : nullptr;
        k_layer_pre<<<SEQ, 256, 0, stream>>>(x, x0, skip, skip_w, l - 6, blk_lam, l, xn_bf);

        gemm(xn_bf, wq_bf + (size_t)l * DIM * DIM, qf, SEQ, DIM, DIM, true);
        gemm(xn_bf, wk_bf + (size_t)l * DIM * DIM, kf, SEQ, DIM, DIM, true);
        gemm(xn_bf, wv_bf + (size_t)l * DIM * DIM, vf, SEQ, DIM, DIM, true);

        k_rope<<<SEQ, 256, 0, stream>>>(qf, qb);
        k_rope<<<SEQ, 256, 0, stream>>>(kf, kb);
        k_vcomb<<<EW, 256, 0, stream>>>(vf, ve + (size_t)VE_IDX[l] * VOC * DIM,
                                        inputs, attn_lam, l, vb);

        gemm(qb, kb, sc, SEQ, SEQ, DIM, true);                    // scores (scale in softmax)
        k_softmax<<<SEQ, 256, 0, stream>>>(sc, docs, dlp, dhp, swnb, pb);
        gemm(pb, vb, yf, SEQ, DIM, SEQ, false);                   // y = p @ v

        cvt(yf, yb, TD);
        gemm(yb, wo_bf + (size_t)l * DIM * DIM, of, SEQ, DIM, DIM, true);
        k_add<<<EW, 256, 0, stream>>>(x, of, TD);

        k_rms_bf<<<SEQ, 256, 0, stream>>>(x, xn_bf);
        gemm(xn_bf, wfc_bf + (size_t)l * FF * DIM, hf, SEQ, FF, DIM, true);
        k_relusq<<<EW * 4, 256, 0, stream>>>(hf, hb, TH);
        gemm(hb, wpr_bf + (size_t)l * DIM * FF, mf, SEQ, DIM, FF, true);
        k_add<<<EW, 256, 0, stream>>>(x, mf, TD);

        if (l < 6) k_copy<<<EW, 256, 0, stream>>>(skips + (size_t)l * TD, x, TD);
    }

    // final norm + logits into d_out, tanh cap, loss
    float* logits = (float*)d_out;
    k_rms_bf<<<SEQ, 256, 0, stream>>>(x, xn_bf);
    gemm(xn_bf, lm_bf, logits, SEQ, VOC, DIM, true);
    k_cap<<<32768, 256, 0, stream>>>(logits, (size_t)SEQ * VOC);
    k_row_loss<<<SEQ, 256, 0, stream>>>(logits, targets, rowls);
    k_loss<<<1, 256, 0, stream>>>(rowls, logits + (size_t)SEQ * VOC);
}
